// FlashMaskAttention_41815801594387
// MI455X (gfx1250) — compile-verified, matrix operand order checked by reading, not run on hardware
//
#include <hip/hip_runtime.h>

typedef __attribute__((ext_vector_type(16))) _Float16 v16h;
typedef __attribute__((ext_vector_type(8)))  _Float16 v8h;
typedef __attribute__((ext_vector_type(8)))  float    v8f;
typedef __attribute__((ext_vector_type(4)))  int      v4i;

#define SQ_  2048
#define SK_  2048
#define H_   16
#define HKV_ 4
#define D_   128
#define GQA_ (H_ / HKV_)

#if __has_builtin(__builtin_amdgcn_global_load_async_to_lds_b128) && \
    __has_builtin(__builtin_amdgcn_s_wait_asynccnt)
#define USE_ASYNC_LDS 1
typedef __attribute__((address_space(1))) v4i g_v4i;   // global b128
typedef __attribute__((address_space(3))) v4i l_v4i;   // LDS b128
#else
#define USE_ASYNC_LDS 0
#endif

// Copy 16 bytes global(f16) -> LDS. Async (ASYNCcnt) when the gfx1250 builtin
// exists, else a b128 load + ds_store_b128 pair.
__device__ __forceinline__ void copy16_g2l(const _Float16* g, _Float16* l) {
#if USE_ASYNC_LDS
    __builtin_amdgcn_global_load_async_to_lds_b128((g_v4i*)g, (l_v4i*)l, 0, 0);
#else
    *(v8h*)l = *(const v8h*)g;
#endif
}

// ---------------- one-time K/V convert kernel ----------------
// kh : [HKV][SK][128] f16 (row-major, same as source minus head interleave)
// vth: [HKV][128][SK] f16 (transposed so PV B-operand reads are contiguous)
__global__ __launch_bounds__(256) void cvt_kv_kernel(
        const float* __restrict__ kg, const float* __restrict__ vg,
        _Float16* __restrict__ kh, _Float16* __restrict__ vth)
{
    const int gid = blockIdx.x * 256 + threadIdx.x;
    const int NK  = HKV_ * SK_ * D_ / 8;          // 131072 vec8 jobs for K
    if (gid < NK) {
        const int dseg = gid & (D_ / 8 - 1);      // 0..15
        const int hs   = gid >> 4;                // hkv*SK + s
        const int s    = hs & (SK_ - 1);
        const int hkv  = hs >> 11;
        const float4* s4 = (const float4*)(kg + ((size_t)s * HKV_ + hkv) * D_ + dseg * 8);
        float4 a = s4[0], b = s4[1];
        v8h o;
        o[0]=(_Float16)a.x; o[1]=(_Float16)a.y; o[2]=(_Float16)a.z; o[3]=(_Float16)a.w;
        o[4]=(_Float16)b.x; o[5]=(_Float16)b.y; o[6]=(_Float16)b.z; o[7]=(_Float16)b.w;
        *(v8h*)(kh + (size_t)hs * D_ + dseg * 8) = o;
    } else {
        const int t    = gid - NK;
        const int sseg = t & (SK_ / 8 - 1);       // 0..255
        const int hd   = t >> 8;                  // hkv*128 + d
        const int d    = hd & (D_ - 1);
        const int hkv  = hd >> 7;
        const int s0   = sseg * 8;
        v8h o;
        #pragma unroll
        for (int i = 0; i < 8; ++i)
            o[i] = (_Float16)vg[((size_t)(s0 + i) * HKV_ + hkv) * D_ + d];
        *(v8h*)(vth + (size_t)hd * SK_ + s0) = o;
    }
}

// ---------------- main flash kernel ----------------
template <bool WS>
__launch_bounds__(128, 1)
__global__ void flashmask_fwd(const float* __restrict__ qg,
                              const float* __restrict__ kg,
                              const float* __restrict__ vg,
                              const int*   __restrict__ sr,
                              const _Float16* __restrict__ kh,
                              const _Float16* __restrict__ vth,
                              float* __restrict__ out)
{
    __shared__ _Float16 Kl[32][128];   // K tile, row-major
    __shared__ _Float16 Vt[128][40];   // V tile, transposed (pad->80B rows)
    __shared__ _Float16 Pl[4][16][32]; // per-wave P transpose scratch

    const int tid  = threadIdx.x;
    const int wave = tid >> 5;
    const int lane = tid & 31;
    const int ln16 = lane & 15;
    const int hi8  = (lane >> 4) * 8;

    const int h      = blockIdx.y;
    const int hkv    = h / GQA_;
    const int q_base = blockIdx.x * 64 + wave * 16;

    // fold 1/sqrt(D) * log2(e) into Q: softmax runs in base-2 domain
    const float QSC = 0.08838834764831845f * 1.44269504088896340f;

    // ---- Q tile -> WMMA A layout (pre-scaled) ----
    v16h QA[4];
    {
        const float* qrow = qg + ((size_t)(q_base + ln16) * H_ + h) * D_;
        #pragma unroll
        for (int c = 0; c < 4; ++c) {
            const int dlo = c * 32 + hi8;
            const int dhi = c * 32 + 16 + hi8;
            #pragma unroll
            for (int i = 0; i < 8; ++i) {
                QA[c][i]     = (_Float16)(qrow[dlo + i] * QSC);
                QA[c][i + 8] = (_Float16)(qrow[dhi + i] * QSC);
            }
        }
    }

    v16h ONES;
    #pragma unroll
    for (int i = 0; i < 16; ++i) ONES[i] = (_Float16)1.0f;

    v8f O[8];
    #pragma unroll
    for (int j = 0; j < 8; ++j) O[j] = v8f{};
    v8f mrow, lrow;
    #pragma unroll
    for (int r = 0; r < 8; ++r) { mrow[r] = -1e30f; lrow[r] = 0.0f; }

    const int nkb  = 2 * blockIdx.x + 2;   // causal truncation
    const int crow = tid >> 2;             // 0..31
    const int cq   = tid & 3;              // 64B segment id

    for (int kb = 0; kb < nkb; ++kb) {
        const int k0 = kb * 32;

        // ---------- stage K/V tiles into LDS ----------
        if constexpr (WS) {
            const _Float16* kp = kh  + ((size_t)(hkv * SK_ + k0 + crow)) * D_ + cq * 32;
            const _Float16* vp = vth + ((size_t)(hkv * D_ + tid)) * SK_ + k0;
            #pragma unroll
            for (int i = 0; i < 4; ++i)
                copy16_g2l(kp + i * 8, &Kl[crow][cq * 32 + i * 8]);
            #pragma unroll
            for (int i = 0; i < 4; ++i)
                copy16_g2l(vp + i * 8, &Vt[tid][i * 8]);
            if (kb + 1 < nkb) {            // gfx1250 global_prefetch_b8
                __builtin_prefetch(kp + 32 * D_, 0, 0);
                __builtin_prefetch(vp + 32, 0, 0);
            }
#if USE_ASYNC_LDS
            __builtin_amdgcn_s_wait_asynccnt(0);
#endif
        } else {
            const float* kp = kg + ((size_t)(k0 + crow) * HKV_ + hkv) * D_ + cq * 32;
            const float* vp = vg + ((size_t)(k0 + crow) * HKV_ + hkv) * D_ + cq * 32;
            const float4* kp4 = (const float4*)kp;
            const float4* vp4 = (const float4*)vp;
            #pragma unroll
            for (int i = 0; i < 8; ++i) {
                float4 f = kp4[i];
                Kl[crow][cq * 32 + 4*i + 0] = (_Float16)f.x;
                Kl[crow][cq * 32 + 4*i + 1] = (_Float16)f.y;
                Kl[crow][cq * 32 + 4*i + 2] = (_Float16)f.z;
                Kl[crow][cq * 32 + 4*i + 3] = (_Float16)f.w;
            }
            #pragma unroll
            for (int i = 0; i < 8; ++i) {
                float4 f = vp4[i];
                Vt[cq * 32 + 4*i + 0][crow] = (_Float16)f.x;
                Vt[cq * 32 + 4*i + 1][crow] = (_Float16)f.y;
                Vt[cq * 32 + 4*i + 2][crow] = (_Float16)f.z;
                Vt[cq * 32 + 4*i + 3][crow] = (_Float16)f.w;
            }
            if (kb + 1 < nkb) {
                __builtin_prefetch(kp + 32 * HKV_ * D_, 0, 0);
                __builtin_prefetch(vp + 32 * HKV_ * D_, 0, 0);
            }
        }
        __syncthreads();

        // ---------- S = (Q*qsc) K^T : two 16x16 tiles ----------
        v8f S[2];
        #pragma unroll
        for (int t = 0; t < 2; ++t) {
            v8f acc = v8f{};
            #pragma unroll
            for (int c = 0; c < 4; ++c) {
                v8h lo = *(const v8h*)&Kl[t * 16 + ln16][c * 32 + hi8];
                v8h hi = *(const v8h*)&Kl[t * 16 + ln16][c * 32 + 16 + hi8];
                v16h bk = __builtin_shufflevector(lo, hi,
                    0,1,2,3,4,5,6,7,8,9,10,11,12,13,14,15);
                acc = __builtin_amdgcn_wmma_f32_16x16x32_f16(
                        false, QA[c], false, bk, (short)0, acc, false, false);
            }
            S[t] = acc;
        }

        // ---------- mask: causal | band ----------
        #pragma unroll
        for (int t = 0; t < 2; ++t) {
            const int col  = k0 + t * 16 + ln16;
            const int sidx = (hkv * SK_ + col) * 2;
            const int cs   = sr[sidx];
            const unsigned bw = (unsigned)(sr[sidx + 1] - cs);
            #pragma unroll
            for (int r = 0; r < 8; ++r) {
                const int row = q_base + r + hi8;
                const bool msk = (row < col) || ((unsigned)(row - cs) < bw);
                S[t][r] = msk ? -1e30f : S[t][r];
            }
        }

        // ---------- joint row-max over both tiles, one 4-step tree ----------
        v8f rm;
        #pragma unroll
        for (int r = 0; r < 8; ++r) rm[r] = fmaxf(S[0][r], S[1][r]);
        #pragma unroll
        for (int xd = 1; xd < 16; xd <<= 1) {
            #pragma unroll
            for (int r = 0; r < 8; ++r)
                rm[r] = fmaxf(rm[r], __shfl_xor(rm[r], xd, 32));
        }
        v8f mnew, alpha;
        #pragma unroll
        for (int r = 0; r < 8; ++r) mnew[r]  = fmaxf(mrow[r], rm[r]);
        #pragma unroll
        for (int r = 0; r < 8; ++r) alpha[r] = exp2f(mrow[r] - mnew[r]);
        mrow = mnew;
        #pragma unroll
        for (int j = 0; j < 8; ++j)
            #pragma unroll
            for (int r = 0; r < 8; ++r) O[j][r] *= alpha[r];

        // ---------- P = exp2(S - m), D-layout -> LDS (f16) ----------
        #pragma unroll
        for (int t = 0; t < 2; ++t) {
            #pragma unroll
            for (int r = 0; r < 8; ++r) {
                float p = exp2f(S[t][r] - mnew[r]);
                Pl[wave][r + hi8][t * 16 + ln16] = (_Float16)p;
            }
        }

        // ---------- P back in A layout (wave-private, DS in-order) ----------
        v16h PA;
        {
            v8h lo = *(const v8h*)&Pl[wave][ln16][hi8];
            v8h hi = *(const v8h*)&Pl[wave][ln16][16 + hi8];
            PA = __builtin_shufflevector(lo, hi,
                0,1,2,3,4,5,6,7,8,9,10,11,12,13,14,15);
        }

        // ---------- row-sum via WMMA: P x ones lands sums in lrow layout ----------
        {
            v8f rs = __builtin_amdgcn_wmma_f32_16x16x32_f16(
                       false, PA, false, ONES, (short)0, v8f{}, false, false);
            #pragma unroll
            for (int r = 0; r < 8; ++r) lrow[r] = lrow[r] * alpha[r] + rs[r];
        }

        // ---------- O += P * V (8 d-tiles) ----------
        #pragma unroll
        for (int j = 0; j < 8; ++j) {
            v8h lo = *(const v8h*)&Vt[j * 16 + ln16][hi8];
            v8h hi = *(const v8h*)&Vt[j * 16 + ln16][16 + hi8];
            v16h bv = __builtin_shufflevector(lo, hi,
                0,1,2,3,4,5,6,7,8,9,10,11,12,13,14,15);
            O[j] = __builtin_amdgcn_wmma_f32_16x16x32_f16(
                     false, PA, false, bv, (short)0, O[j], false, false);
        }
        __syncthreads();
    }

    // ---------- normalize + store ----------
    float inv[8];
    #pragma unroll
    for (int r = 0; r < 8; ++r) inv[r] = 1.0f / fmaxf(lrow[r], 1e-20f);
    #pragma unroll
    for (int j = 0; j < 8; ++j) {
        #pragma unroll
        for (int r = 0; r < 8; ++r) {
            const int row = q_base + r + hi8;
            const int d   = j * 16 + ln16;
            out[((size_t)row * H_ + h) * D_ + d] = O[j][r] * inv[r];
        }
    }
}

extern "C" void kernel_launch(void* const* d_in, const int* in_sizes, int n_in,
                              void* d_out, int out_size, void* d_ws, size_t ws_size,
                              hipStream_t stream) {
    const float* q  = (const float*)d_in[0];
    const float* k  = (const float*)d_in[1];
    const float* v  = (const float*)d_in[2];
    const int*   sr = (const int*)d_in[3];
    float* out = (float*)d_out;
    (void)in_sizes; (void)n_in; (void)out_size;

    dim3 grid(SQ_ / 64, H_, 1);
    dim3 block(128, 1, 1);

    const size_t kv_elems = (size_t)HKV_ * SK_ * D_;          // 1M f16 each
    const size_t need     = kv_elems * sizeof(_Float16) * 2;  // 4 MB
    if (ws_size >= need) {
        _Float16* kh  = (_Float16*)d_ws;
        _Float16* vth = kh + kv_elems;
        cvt_kv_kernel<<<(2 * (int)(kv_elems / 8)) / 256, 256, 0, stream>>>(k, v, kh, vth);
        flashmask_fwd<true><<<grid, block, 0, stream>>>(q, k, v, sr, kh, vth, out);
    } else {
        flashmask_fwd<false><<<grid, block, 0, stream>>>(q, k, v, sr, nullptr, nullptr, out);
    }
}
